// VLLMSSESWAGLA_73976516706668
// MI455X (gfx1250) — matrix operand order repaired and checked by reading, not compiled, not run on hardware
//
#include <hip/hip_runtime.h>
#include <hip/hip_bf16.h>
#include <math.h>
#include <stdint.h>

// ---------------------------------------------------------------------------
// Model dimensions (compile-time; match the reference)
// ---------------------------------------------------------------------------
#define T_SEQ   2048
#define DMODEL  2048
#define NH      6
#define DKH     256
#define DVH     256
#define HD      (NH * DKH)   // 1536
#define NCHUNK  32
#define CHUNK_T 64
#define WINDOW  1024
#define VT_LD   72           // padded LDS stride (multiple of 8 halves -> 16B aligned frags)

typedef _Float16 v16h __attribute__((ext_vector_type(16)));
typedef _Float16 v8h  __attribute__((ext_vector_type(8)));
typedef float    v8f  __attribute__((ext_vector_type(8)));
typedef int      v4i  __attribute__((ext_vector_type(4)));

typedef __attribute__((address_space(1))) v4i* gas_v4i;
typedef __attribute__((address_space(3))) v4i* las_v4i;

// ---------------------------------------------------------------------------
// CDNA5 async global->LDS copy (ASYNCcnt path, CDNA5 ISA)
// Guarded: falls back to direct LDS stores if the builtin is absent.
// ---------------------------------------------------------------------------
#if __has_builtin(__builtin_amdgcn_global_load_async_to_lds_b128)
#define HAVE_ASYNC_LDS 1
#endif

__device__ __forceinline__ void cp_async16(_Float16* lds, const _Float16* g) {
#ifdef HAVE_ASYNC_LDS
    // AS3 pointer = low 32 bits of the generic LDS address (aperture stripped)
    __builtin_amdgcn_global_load_async_to_lds_b128(
        (gas_v4i)(uintptr_t)g,
        (las_v4i)(uint32_t)(uintptr_t)lds,
        0, 0);
#else
    *(v8h*)lds = *(const v8h*)g;
#endif
}

__device__ __forceinline__ void cp_async_wait_all() {
#ifdef HAVE_ASYNC_LDS
#if __has_builtin(__builtin_amdgcn_s_wait_asynccnt)
    __builtin_amdgcn_s_wait_asynccnt(0);
#else
    asm volatile("s_wait_asynccnt 0x0" ::: "memory");
#endif
#endif
}

// ---------------------------------------------------------------------------
// WMMA fragment helpers (wave32; layouts per CDNA5 ISA §7.12.2)
// ---------------------------------------------------------------------------
__device__ __forceinline__ v16h frag_a(const _Float16* A, int lda) {
    int lane = threadIdx.x & 31;
    int row  = lane & 15;
    int ko   = (lane < 16) ? 0 : 8;
    const _Float16* p = A + (size_t)row * lda + ko;
    v8h lo = *(const v8h*)(p);
    v8h hi = *(const v8h*)(p + 16);
    v16h r;
#pragma unroll
    for (int i = 0; i < 8; ++i) { r[i] = lo[i]; r[i + 8] = hi[i]; }
    return r;
}

__device__ __forceinline__ v16h frag_b(const _Float16* Bt, int ldb) {
    int lane = threadIdx.x & 31;
    int col  = lane & 15;
    int ko   = (lane < 16) ? 0 : 8;
    const _Float16* p = Bt + (size_t)col * ldb + ko;
    v8h lo = *(const v8h*)(p);
    v8h hi = *(const v8h*)(p + 16);
    v16h r;
#pragma unroll
    for (int i = 0; i < 8; ++i) { r[i] = lo[i]; r[i + 8] = hi[i]; }
    return r;
}

__device__ __forceinline__ v8f wmma_f16(v16h a, v16h b, v8f c) {
    return __builtin_amdgcn_wmma_f32_16x16x32_f16(false, a, false, b, (short)0, c,
                                                  false, false);
}

__device__ __forceinline__ void store_c(float* C, int ldc, v8f c) {
    int lane  = threadIdx.x & 31;
    int col   = lane & 15;
    int rbase = (lane < 16) ? 0 : 8;
#pragma unroll
    for (int i = 0; i < 8; ++i) C[(size_t)(rbase + i) * ldc + col] = c[i];
}

// ---------------------------------------------------------------------------
// Block WMMA GEMM: C[M,N] = A[M,K](fp16,row) x Bt[N,K](fp16,row)
// grid (N/128, M/128), block 256 (8 waves). 128x128 block tile, K sliced by 32,
// double-buffered LDS staged with async copies; wave computes 32x64 (8 WMMA/step).
// ---------------------------------------------------------------------------
__global__ void __launch_bounds__(256) k_gemm(const _Float16* __restrict__ A,
                                              const _Float16* __restrict__ Bt,
                                              float* __restrict__ C,
                                              int M, int N, int K) {
    __shared__ alignas(16) _Float16 As[2][128 * 32];
    __shared__ alignas(16) _Float16 Bs[2][128 * 32];
    int m0 = blockIdx.y * 128;
    int n0 = blockIdx.x * 128;
    int tid = threadIdx.x, wave = tid >> 5;
    int wm = (wave >> 1) * 32;   // wave row offset within block tile
    int wn = (wave & 1) * 64;    // wave col offset within block tile
    v8f acc[2][4] = {};

    auto stage = [&](int buf, int k0) {
#pragma unroll
        for (int r = 0; r < 2; ++r) {
            int chunk = tid + r * 256;       // 512 x 16B chunks per 8KB tile
            int row   = chunk >> 2;
            int co    = (chunk & 3) * 8;
            cp_async16(&As[buf][row * 32 + co], A  + (size_t)(m0 + row) * K + k0 + co);
            cp_async16(&Bs[buf][row * 32 + co], Bt + (size_t)(n0 + row) * K + k0 + co);
        }
    };

    stage(0, 0);
    cp_async_wait_all();
    __syncthreads();
    int buf = 0;
    for (int k0 = 0; k0 < K; k0 += 32) {
        if (k0 + 32 < K) stage(buf ^ 1, k0 + 32);   // overlap next slice
        v16h a0 = frag_a(&As[buf][(wm + 0)  * 32], 32);
        v16h a1 = frag_a(&As[buf][(wm + 16) * 32], 32);
#pragma unroll
        for (int j = 0; j < 4; ++j) {
            v16h b = frag_b(&Bs[buf][(wn + j * 16) * 32], 32);
            acc[0][j] = wmma_f16(a0, b, acc[0][j]);
            acc[1][j] = wmma_f16(a1, b, acc[1][j]);
        }
        cp_async_wait_all();
        __syncthreads();
        buf ^= 1;
    }
#pragma unroll
    for (int i = 0; i < 2; ++i)
#pragma unroll
        for (int j = 0; j < 4; ++j)
            store_c(C + (size_t)(m0 + wm + i * 16) * N + n0 + wn + j * 16, N, acc[i][j]);
}

// ---------------------------------------------------------------------------
// Naive fp32 GEMM for tiny shapes (K=16 or N<=16). mode 1: logsigmoid/16 epilogue
// ---------------------------------------------------------------------------
__global__ void k_gemm_naive(const float* __restrict__ A, const float* __restrict__ B,
                             const float* __restrict__ bias, float* __restrict__ C,
                             int M, int N, int K, int mode) {
    int idx = blockIdx.x * blockDim.x + threadIdx.x;
    if (idx >= M * N) return;
    int m = idx / N, n = idx - m * N;
    float acc = bias ? bias[n] : 0.0f;
    for (int k = 0; k < K; ++k) acc += A[(size_t)m * K + k] * B[(size_t)k * N + n];
    if (mode == 1) {
        float z = acc;
        acc = (fminf(z, 0.0f) - log1pf(expf(-fabsf(z)))) * (1.0f / 16.0f);
    }
    C[idx] = acc;
}

// ---------------------------------------------------------------------------
// Elementwise / conversion kernels
// ---------------------------------------------------------------------------
__global__ void k_cvt_fp16(const float* __restrict__ s, _Float16* __restrict__ d, size_t n) {
    size_t i = (size_t)blockIdx.x * blockDim.x + threadIdx.x;
    if (i < n) d[i] = (_Float16)s[i];
}

__global__ void k_fill_zero(float* __restrict__ p, size_t n) {
    size_t i = (size_t)blockIdx.x * blockDim.x + threadIdx.x;
    if (i < n) p[i] = 0.0f;
}

__global__ void k_ewmul(float* __restrict__ a, const float* __restrict__ b, size_t n) {
    size_t i = (size_t)blockIdx.x * blockDim.x + threadIdx.x;
    if (i < n) a[i] *= b[i];
}

// W[K,N] fp32 -> Wt[N,K] fp16 (tiled transpose)
__global__ void k_transpose_cvt(const float* __restrict__ W, _Float16* __restrict__ Wt,
                                int K, int N) {
    __shared__ float tile[32][33];
    int kb = blockIdx.y * 32, nb = blockIdx.x * 32;
    int tx = threadIdx.x & 31, ty = threadIdx.x >> 5;
    for (int i = ty; i < 32; i += 8) {
        int kk = kb + i, nn = nb + tx;
        tile[i][tx] = (kk < K && nn < N) ? W[(size_t)kk * N + nn] : 0.0f;
    }
    __syncthreads();
    for (int i = ty; i < 32; i += 8) {
        int nn = nb + i, kk = kb + tx;
        if (nn < N && kk < K) Wt[(size_t)nn * K + kk] = (_Float16)tile[tx][i];
    }
}

// softmax over contiguous rows of 256 (k projection), in place
__global__ void k_softmax256(float* __restrict__ p) {
    __shared__ float red[256];
    size_t off = (size_t)blockIdx.x * 256;
    int tid = threadIdx.x;
    float x = p[off + tid];
    red[tid] = x; __syncthreads();
    for (int s = 128; s > 0; s >>= 1) {
        if (tid < s) red[tid] = fmaxf(red[tid], red[tid + s]);
        __syncthreads();
    }
    float mx = red[0]; __syncthreads();
    float e = expf(x - mx);
    red[tid] = e; __syncthreads();
    for (int s = 128; s > 0; s >>= 1) {
        if (tid < s) red[tid] += red[tid + s];
        __syncthreads();
    }
    p[off + tid] = e / red[0];
}

// top-1 routing over P=4 experts
__global__ void k_argmax4(const float* __restrict__ e, int* __restrict__ sel, int T) {
    int t = blockIdx.x * blockDim.x + threadIdx.x;
    if (t >= T) return;
    const float* p = e + (size_t)t * 4;
    int best = 0; float bv = p[0];
#pragma unroll
    for (int i = 1; i < 4; ++i) { if (p[i] > bv) { bv = p[i]; best = i; } }
    sel[t] = best;
}

// RoPE + fp16 cast: src (t, h*256+d) fp32 -> dst fp16 (rotated)
__global__ void k_rope(const float* __restrict__ src, _Float16* __restrict__ dst) {
    int id = blockIdx.x * blockDim.x + threadIdx.x;
    const int half = DKH / 2;
    if (id >= T_SEQ * NH * half) return;
    int i  = id % half;
    int th = id / half;
    int h  = th % NH;
    int t  = th / NH;
    size_t base = (size_t)t * HD + (size_t)h * DKH;
    float x1 = src[base + i], x2 = src[base + i + half];
    float inv = __powf(10000.0f, -(float)(2 * i) / (float)DKH);
    float fr  = (float)t * inv;
    float s, c;
    __sincosf(fr, &s, &c);
    dst[base + i]        = (_Float16)(x1 * c - x2 * s);
    dst[base + i + half] = (_Float16)(x2 * c + x1 * s);
}

// ---------------------------------------------------------------------------
// GLA prep: per (chunk, head), thread = d. Cumsum gate, emit qi, k*e^{-G} (fp16),
// kd = k*e^{Gt-G}, and Gt.
// ---------------------------------------------------------------------------
__global__ void k_gla_prep(const float* __restrict__ q, const float* __restrict__ k,
                           const float* __restrict__ g, float* __restrict__ qi,
                           float* __restrict__ kd, _Float16* __restrict__ qih,
                           _Float16* __restrict__ keh, float* __restrict__ Gt) {
    int n = blockIdx.x, h = blockIdx.y, d = threadIdx.x;
    size_t base = (size_t)(n * CHUNK_T) * HD + (size_t)h * DKH + d;
    float c = 0.0f;
    for (int t = 0; t < CHUNK_T; ++t) c += g[base + (size_t)t * HD];
    float gt = c;
    Gt[((size_t)h * NCHUNK + n) * DKH + d] = gt;
    c = 0.0f;
    for (int t = 0; t < CHUNK_T; ++t) {
        size_t o = base + (size_t)t * HD;
        c += g[o];
        float qq = q[o], kk = k[o];
        float qiv = qq * expf(c);
        qi[o]  = qiv;
        kd[o]  = kk * expf(gt - c);
        qih[o] = (_Float16)qiv;
        keh[o] = (_Float16)(kk * expf(-c));
    }
}

// ---------------------------------------------------------------------------
// GLA intra-chunk: A = qi ke^T (WMMA), mask (causal & same-route), o = A V (WMMA)
// grid (NCHUNK, NH), block 256 (8 waves). Static LDS ~60.5 KB.
// ---------------------------------------------------------------------------
__global__ void k_gla_intra(const _Float16* __restrict__ qih,
                            const _Float16* __restrict__ keh,
                            const float* __restrict__ v, const int* __restrict__ sel,
                            float* __restrict__ o) {
    __shared__ float    Ss[64 * 64];
    __shared__ alignas(16) _Float16 Ps[64 * 64];
    __shared__ alignas(16) _Float16 Vt[256 * VT_LD];
    __shared__ int selc[64];
    int n = blockIdx.x, h = blockIdx.y;
    int tid = threadIdx.x, wave = tid >> 5;
    int t0 = n * CHUNK_T;

    if (tid < 64) selc[tid] = sel[t0 + tid];
    for (int idx = tid; idx < 64 * 256; idx += 256) {
        int s = idx >> 8, e = idx & 255;
        Vt[e * VT_LD + s] = (_Float16)v[(size_t)(t0 + s) * HD + (size_t)h * DKH + e];
    }
    __syncthreads();

    { // scores: 4x4 tiles, 2 per wave
        int mt  = wave >> 1;
        int stb = (wave & 1) * 2;
        v8f a0 = {}, a1 = {};
        const _Float16* Ab = qih + (size_t)(t0 + mt * 16) * HD + (size_t)h * DKH;
        const _Float16* B0 = keh + (size_t)(t0 + (stb + 0) * 16) * HD + (size_t)h * DKH;
        const _Float16* B1 = keh + (size_t)(t0 + (stb + 1) * 16) * HD + (size_t)h * DKH;
        for (int k0 = 0; k0 < DKH; k0 += 32) {
            v16h a = frag_a(Ab + k0, HD);
            a0 = wmma_f16(a, frag_b(B0 + k0, HD), a0);
            a1 = wmma_f16(a, frag_b(B1 + k0, HD), a1);
        }
        store_c(Ss + (mt * 16) * 64 + (stb + 0) * 16, 64, a0);
        store_c(Ss + (mt * 16) * 64 + (stb + 1) * 16, 64, a1);
    }
    __syncthreads();

    for (int idx = tid; idx < 64 * 64; idx += 256) {
        int r = idx >> 6, s2 = idx & 63;
        bool keep = (s2 <= r) && (selc[r] == selc[s2]);
        Ps[idx] = (_Float16)(keep ? Ss[idx] : 0.0f);
    }
    __syncthreads();

    { // o_intra = P(64x64) x V(64x256): wave owns 1 M-tile x 8 N-tiles
        int mt  = wave & 3;
        int ntb = (wave >> 2) * 8;
        v8f acc[8] = {};
#pragma unroll
        for (int ks = 0; ks < 2; ++ks) {
            v16h a = frag_a((const _Float16*)Ps + (mt * 16) * 64 + ks * 32, 64);
#pragma unroll
            for (int j = 0; j < 8; ++j) {
                v16h b = frag_b((const _Float16*)Vt + ((ntb + j) * 16) * VT_LD + ks * 32,
                                VT_LD);
                acc[j] = wmma_f16(a, b, acc[j]);
            }
        }
#pragma unroll
        for (int j = 0; j < 8; ++j)
            store_c(o + (size_t)(t0 + mt * 16) * HD + (size_t)h * DKH + (ntb + j) * 16,
                    HD, acc[j]);
    }
}

// ---------------------------------------------------------------------------
// GLA inter-chunk sequential scan. grid (NH, 4), block 64; thread owns state
// column e exclusively -> no synchronization needed. S: [h][p][d][e] fp32.
// ---------------------------------------------------------------------------
__global__ void k_gla_inter(const float* __restrict__ qi, const float* __restrict__ kd,
                            const float* __restrict__ v, const float* __restrict__ Gt,
                            const int* __restrict__ sel, float* __restrict__ S,
                            float* __restrict__ o) {
    int h = blockIdx.x;
    int e = blockIdx.y * 64 + threadIdx.x;
    float* Sh = S + (size_t)h * 4 * DKH * DVH;
    for (int n = 0; n < NCHUNK; ++n) {
        int t0 = n * CHUNK_T;
        // o_inter (uses state BEFORE update)
        for (int t = 0; t < CHUNK_T; ++t) {
            int p = sel[t0 + t];
            const float* qrow = qi + (size_t)(t0 + t) * HD + (size_t)h * DKH;
            const float* Sp   = Sh + (size_t)p * DKH * DVH + e;
            float acc = 0.0f;
#pragma unroll 4
            for (int d = 0; d < DKH; ++d) acc += qrow[d] * Sp[(size_t)d * DVH];
            o[(size_t)(t0 + t) * HD + (size_t)h * DKH + e] += acc;
        }
        // decay all routed states
        const float* gtp = Gt + ((size_t)h * NCHUNK + n) * DKH;
        for (int p = 0; p < 4; ++p) {
            float* Sp = Sh + (size_t)p * DKH * DVH + e;
#pragma unroll 4
            for (int d = 0; d < DKH; ++d) Sp[(size_t)d * DVH] *= expf(gtp[d]);
        }
        // rank-1 updates for routed tokens
        for (int t = 0; t < CHUNK_T; ++t) {
            int p = sel[t0 + t];
            float vt = v[(size_t)(t0 + t) * HD + (size_t)h * DKH + e];
            const float* krow = kd + (size_t)(t0 + t) * HD + (size_t)h * DKH;
            float* Sp = Sh + (size_t)p * DKH * DVH + e;
#pragma unroll 4
            for (int d = 0; d < DKH; ++d) Sp[(size_t)d * DVH] += krow[d] * vt;
        }
    }
}

// ---------------------------------------------------------------------------
// RMSNorm(o, w_onorm) * silu(gate) -> fp16, per (t,h) row of 256
// ---------------------------------------------------------------------------
__global__ void k_onorm_gate(const float* __restrict__ o, const float* __restrict__ gate,
                             const float* __restrict__ w, _Float16* __restrict__ oh) {
    __shared__ float red[256];
    int tid = threadIdx.x;
    size_t off = (size_t)blockIdx.x * 256;
    float x = o[off + tid];
    red[tid] = x * x; __syncthreads();
    for (int s = 128; s > 0; s >>= 1) {
        if (tid < s) red[tid] += red[tid + s];
        __syncthreads();
    }
    float inv = rsqrtf(red[0] * (1.0f / 256.0f) + 1e-5f);
    float gv = gate[off + tid];
    float silu = gv / (1.0f + expf(-gv));
    oh[off + tid] = (_Float16)(x * inv * w[tid] * silu);
}

// ---------------------------------------------------------------------------
// Sliding-window flash attention (WMMA QK^T and P.V, online softmax).
// grid (T/64, NH), block 256 (8 waves). Static LDS ~62 KB.
// ---------------------------------------------------------------------------
__global__ void k_swa_flash(const _Float16* __restrict__ qh, const _Float16* __restrict__ kh,
                            const _Float16* __restrict__ vh, _Float16* __restrict__ outh) {
    __shared__ float    Ss[64 * 64];
    __shared__ alignas(16) _Float16 Ps[64 * 64];
    __shared__ alignas(16) _Float16 Vt[256 * VT_LD];
    __shared__ float rowm[64], rowl[64], rowa[64];
    int qt = blockIdx.x, h = blockIdx.y;
    int tid = threadIdx.x, lane = tid & 31, wave = tid >> 5;
    int m0 = qt * 64;
    int mt = wave & 3, ntb = (wave >> 2) * 8;   // output tile ownership
    v8f acc[8] = {};
    if (tid < 64) { rowm[tid] = -3.0e38f; rowl[tid] = 0.0f; }
    __syncthreads();

    const float scale = 0.0625f;                // 1/sqrt(256)
    int st_lo = (qt >= 16) ? (qt - 16) : 0;     // WINDOW/64 = 16 tiles back
    for (int st = st_lo; st <= qt; ++st) {
        int s0 = st * 64;
        for (int idx = tid; idx < 64 * 256; idx += 256) {
            int s = idx >> 8, e = idx & 255;
            Vt[e * VT_LD + s] = vh[(size_t)(s0 + s) * HD + (size_t)h * DKH + e];
        }
        __syncthreads();                         // A: Vt staged
        { // scores: 16 tiles, 2 per wave
            int mts = wave >> 1;
            int stb = (wave & 1) * 2;
            v8f a0 = {}, a1 = {};
            const _Float16* Ab = qh + (size_t)(m0 + mts * 16) * HD + (size_t)h * DKH;
            const _Float16* B0 = kh + (size_t)(s0 + stb * 16) * HD + (size_t)h * DKH;
            const _Float16* B1 = kh + (size_t)(s0 + (stb + 1) * 16) * HD + (size_t)h * DKH;
            for (int k0 = 0; k0 < DKH; k0 += 32) {
                v16h a = frag_a(Ab + k0, HD);
                a0 = wmma_f16(a, frag_b(B0 + k0, HD), a0);
                a1 = wmma_f16(a, frag_b(B1 + k0, HD), a1);
            }
            store_c(Ss + (mts * 16) * 64 + stb * 16, 64, a0);
            store_c(Ss + (mts * 16) * 64 + (stb + 1) * 16, 64, a1);
        }
        __syncthreads();                         // B: Ss ready
        if (tid < 64) {                          // online softmax per row
            int r = tid, tg = m0 + r;
            float mold = rowm[r];
            float mnew = mold;
            for (int s2 = 0; s2 < 64; ++s2) {
                int sg = s0 + s2;
                if (sg <= tg && tg - sg < WINDOW)
                    mnew = fmaxf(mnew, Ss[r * 64 + s2] * scale);
            }
            float alpha = expf(mold - mnew);
            float lsum = 0.0f;
            for (int s2 = 0; s2 < 64; ++s2) {
                int sg = s0 + s2;
                float pe = 0.0f;
                if (sg <= tg && tg - sg < WINDOW)
                    pe = expf(Ss[r * 64 + s2] * scale - mnew);
                Ps[r * 64 + s2] = (_Float16)pe;
                lsum += pe;
            }
            rowa[r] = alpha;
            rowm[r] = mnew;
            rowl[r] = rowl[r] * alpha + lsum;
        }
        __syncthreads();                         // C: Ps/rowa ready
        {
            int rbase = (lane < 16) ? 0 : 8;
#pragma unroll
            for (int j = 0; j < 8; ++j) {
                v8f t = acc[j];
#pragma unroll
                for (int i = 0; i < 8; ++i) t[i] *= rowa[mt * 16 + rbase + i];
                acc[j] = t;
            }
#pragma unroll
            for (int ks = 0; ks < 2; ++ks) {
                v16h a = frag_a((const _Float16*)Ps + (mt * 16) * 64 + ks * 32, 64);
#pragma unroll
                for (int j = 0; j < 8; ++j) {
                    v16h b = frag_b((const _Float16*)Vt + ((ntb + j) * 16) * VT_LD + ks * 32,
                                    VT_LD);
                    acc[j] = wmma_f16(a, b, acc[j]);
                }
            }
        }
        __syncthreads();                         // D: done with Vt/Ss/Ps this iter
    }
    { // epilogue: out = acc / l
        int rbase = (lane < 16) ? 0 : 8;
        int col   = lane & 15;
#pragma unroll
        for (int j = 0; j < 8; ++j) {
#pragma unroll
            for (int i = 0; i < 8; ++i) {
                int r = mt * 16 + rbase + i;
                float val = acc[j][i] / rowl[r];
                outh[(size_t)(m0 + r) * HD + (size_t)h * DKH + (ntb + j) * 16 + col] =
                    (_Float16)val;
            }
        }
    }
}

// ---------------------------------------------------------------------------
// Final merge: rmsnorm(sse)*w1 + rmsnorm(swa)*w2 per row of DM
// ---------------------------------------------------------------------------
__global__ void k_merge(const float* __restrict__ a, const float* __restrict__ b,
                        const float* __restrict__ wa, const float* __restrict__ wb,
                        float* __restrict__ out) {
    __shared__ float red[256];
    int tid = threadIdx.x;
    size_t off = (size_t)blockIdx.x * DMODEL;
    float sa = 0.0f, sb = 0.0f;
    for (int i = tid; i < DMODEL; i += 256) {
        float x = a[off + i]; sa += x * x;
        float y = b[off + i]; sb += y * y;
    }
    red[tid] = sa; __syncthreads();
    for (int s = 128; s > 0; s >>= 1) { if (tid < s) red[tid] += red[tid + s]; __syncthreads(); }
    float ia = rsqrtf(red[0] / (float)DMODEL + 1e-5f);
    __syncthreads();
    red[tid] = sb; __syncthreads();
    for (int s = 128; s > 0; s >>= 1) { if (tid < s) red[tid] += red[tid + s]; __syncthreads(); }
    float ib = rsqrtf(red[0] / (float)DMODEL + 1e-5f);
    for (int i = tid; i < DMODEL; i += 256)
        out[off + i] = a[off + i] * ia * wa[i] + b[off + i] * ib * wb[i];
}

// ---------------------------------------------------------------------------
// Host orchestration
// ---------------------------------------------------------------------------
extern "C" void kernel_launch(void* const* d_in, const int* in_sizes, int n_in,
                              void* d_out, int out_size, void* d_ws, size_t ws_size,
                              hipStream_t stream) {
    const float* x      = (const float*)d_in[0];
    const float* Wq     = (const float*)d_in[1];
    const float* Wk     = (const float*)d_in[2];
    const float* Wv     = (const float*)d_in[3];
    const float* Wlq0   = (const float*)d_in[4];
    const float* Wlq1   = (const float*)d_in[5];
    const float* Wlk0   = (const float*)d_in[6];
    const float* Wlk1   = (const float*)d_in[7];
    const float* Wgk0   = (const float*)d_in[8];
    const float* Wgk1   = (const float*)d_in[9];
    const float* bgk1   = (const float*)d_in[10];
    const float* We     = (const float*)d_in[11];
    const float* Wg0    = (const float*)d_in[12];
    const float* Wg1    = (const float*)d_in[13];
    const float* wonorm = (const float*)d_in[14];
    const float* Wsseo  = (const float*)d_in[15];
    const float* Wswaq  = (const float*)d_in[16];
    const float* Wswak  = (const float*)d_in[17];
    const float* Wswav  = (const float*)d_in[18];
    const float* Wswao  = (const float*)d_in[19];
    const float* wsse   = (const float*)d_in[20];
    const float* wswa   = (const float*)d_in[21];
    (void)in_sizes; (void)n_in; (void)out_size; (void)ws_size;

    char* base = (char*)d_ws;
    size_t off = 0;
    auto alloc = [&](size_t bytes) -> void* {
        void* p = base + off;
        off = (off + bytes + 255) & ~(size_t)255;
        return p;
    };
    const size_t TD   = (size_t)T_SEQ * DMODEL;
    const size_t TH   = (size_t)T_SEQ * HD;
    const size_t T256 = (size_t)T_SEQ * DVH;

    // fp16 buffers
    _Float16* xh      = (_Float16*)alloc(TD * 2);
    _Float16* Wq_t    = (_Float16*)alloc((size_t)HD * DMODEL * 2);
    _Float16* Wk_t    = (_Float16*)alloc((size_t)HD * DMODEL * 2);
    _Float16* Wv_t    = (_Float16*)alloc((size_t)HD * DMODEL * 2);
    _Float16* Wlq0_t  = (_Float16*)alloc((size_t)DVH * DMODEL * 2);
    _Float16* Wlq1_t  = (_Float16*)alloc((size_t)HD * DVH * 2);
    _Float16* Wlk0_t  = (_Float16*)alloc((size_t)DVH * DMODEL * 2);
    _Float16* Wlk1_t  = (_Float16*)alloc((size_t)HD * DVH * 2);
    _Float16* Wg0_t   = (_Float16*)alloc((size_t)DVH * DMODEL * 2);
    _Float16* Wg1_t   = (_Float16*)alloc((size_t)HD * DVH * 2);
    _Float16* Wsseo_t = (_Float16*)alloc((size_t)DMODEL * HD * 2);
    _Float16* Wswaq_t = (_Float16*)alloc((size_t)HD * DMODEL * 2);
    _Float16* Wswak_t = (_Float16*)alloc((size_t)HD * DMODEL * 2);
    _Float16* Wswav_t = (_Float16*)alloc((size_t)HD * DMODEL * 2);
    _Float16* Wswao_t = (_Float16*)alloc((size_t)DMODEL * HD * 2);
    _Float16* tmp256h = (_Float16*)alloc(T256 * 2);
    _Float16* qih     = (_Float16*)alloc(TH * 2);
    _Float16* keh     = (_Float16*)alloc(TH * 2);
    _Float16* qsh     = (_Float16*)alloc(TH * 2);
    _Float16* ksh     = (_Float16*)alloc(TH * 2);
    _Float16* vsh     = (_Float16*)alloc(TH * 2);
    _Float16* oh      = (_Float16*)alloc(TH * 2);
    _Float16* swah    = (_Float16*)alloc(TH * 2);
    // fp32 buffers
    float* qbuf    = (float*)alloc(TH * 4);
    float* kbuf    = (float*)alloc(TH * 4);
    float* vbuf    = (float*)alloc(TH * 4);
    float* gbuf    = (float*)alloc(TH * 4);
    float* tmp_ts  = (float*)alloc(TH * 4);
    float* gatebuf = (float*)alloc(TH * 4);
    float* qs      = (float*)alloc(TH * 4);
    float* ks      = (float*)alloc(TH * 4);
    float* vs      = (float*)alloc(TH * 4);
    float* qi_g    = (float*)alloc(TH * 4);
    float* kd_g    = (float*)alloc(TH * 4);
    float* obuf    = (float*)alloc(TH * 4);
    float* sse_out = (float*)alloc(TD * 4);
    float* swa_out = (float*)alloc(TD * 4);
    float* tmp256f = (float*)alloc(T256 * 4);
    float* tmp16   = (float*)alloc((size_t)T_SEQ * 16 * 4);
    float* tmp4    = (float*)alloc((size_t)T_SEQ * 4 * 4);
    float* Gt_g    = (float*)alloc((size_t)NH * NCHUNK * DKH * 4);
    float* Sstate  = (float*)alloc((size_t)NH * 4 * DKH * DVH * 4);
    int*   sel     = (int*)alloc((size_t)T_SEQ * 4);

    auto cvt = [&](const float* s, _Float16* d, size_t n) {
        k_cvt_fp16<<<dim3((unsigned)((n + 255) / 256)), 256, 0, stream>>>(s, d, n);
    };
    auto xpose = [&](const float* W, _Float16* Wt, int K, int N) {
        k_transpose_cvt<<<dim3((N + 31) / 32, (K + 31) / 32), 256, 0, stream>>>(W, Wt, K, N);
    };
    auto gemm = [&](const _Float16* A, const _Float16* Bt, float* C, int M, int N, int K) {
        k_gemm<<<dim3(N / 128, M / 128), 256, 0, stream>>>(A, Bt, C, M, N, K);
    };

    // ---- fp16 staging ----
    cvt(x, xh, TD);
    xpose(Wq,    Wq_t,    DMODEL, HD);
    xpose(Wk,    Wk_t,    DMODEL, HD);
    xpose(Wv,    Wv_t,    DMODEL, HD);
    xpose(Wlq0,  Wlq0_t,  DMODEL, DVH);
    xpose(Wlq1,  Wlq1_t,  DVH,    HD);
    xpose(Wlk0,  Wlk0_t,  DMODEL, DVH);
    xpose(Wlk1,  Wlk1_t,  DVH,    HD);
    xpose(Wg0,   Wg0_t,   DMODEL, DVH);
    xpose(Wg1,   Wg1_t,   DVH,    HD);
    xpose(Wsseo, Wsseo_t, HD,     DMODEL);
    xpose(Wswaq, Wswaq_t, DMODEL, HD);
    xpose(Wswak, Wswak_t, DMODEL, HD);
    xpose(Wswav, Wswav_t, DMODEL, HD);
    xpose(Wswao, Wswao_t, HD,     DMODEL);

    // ---- projections (WMMA) ----
    gemm(xh, Wq_t, qbuf, T_SEQ, HD, DMODEL);
    gemm(xh, Wlq0_t, tmp256f, T_SEQ, DVH, DMODEL);
    cvt(tmp256f, tmp256h, T256);
    gemm(tmp256h, Wlq1_t, tmp_ts, T_SEQ, HD, DVH);
    k_ewmul<<<dim3((unsigned)((TH + 255) / 256)), 256, 0, stream>>>(qbuf, tmp_ts, TH);

    gemm(xh, Wk_t, kbuf, T_SEQ, HD, DMODEL);
    gemm(xh, Wlk0_t, tmp256f, T_SEQ, DVH, DMODEL);
    cvt(tmp256f, tmp256h, T256);
    gemm(tmp256h, Wlk1_t, tmp_ts, T_SEQ, HD, DVH);
    k_ewmul<<<dim3((unsigned)((TH + 255) / 256)), 256, 0, stream>>>(kbuf, tmp_ts, TH);
    k_softmax256<<<T_SEQ * NH, 256, 0, stream>>>(kbuf);

    gemm(xh, Wv_t, vbuf, T_SEQ, HD, DMODEL);

    gemm(xh, Wg0_t, tmp256f, T_SEQ, DVH, DMODEL);
    cvt(tmp256f, tmp256h, T256);
    gemm(tmp256h, Wg1_t, gatebuf, T_SEQ, HD, DVH);

    // gate g and routing (tiny GEMMs, naive fp32)
    k_gemm_naive<<<(T_SEQ * 16 + 255) / 256, 256, 0, stream>>>(x, Wgk0, nullptr, tmp16,
                                                               T_SEQ, 16, DMODEL, 0);
    k_gemm_naive<<<(T_SEQ * HD + 255) / 256, 256, 0, stream>>>(tmp16, Wgk1, bgk1, gbuf,
                                                               T_SEQ, HD, 16, 1);
    k_gemm_naive<<<(T_SEQ * 4 + 255) / 256, 256, 0, stream>>>(x, We, nullptr, tmp4,
                                                              T_SEQ, 4, DMODEL, 0);
    k_argmax4<<<(T_SEQ + 255) / 256, 256, 0, stream>>>(tmp4, sel, T_SEQ);

    // ---- SSE-GLA ----
    k_gla_prep<<<dim3(NCHUNK, NH), 256, 0, stream>>>(qbuf, kbuf, gbuf, qi_g, kd_g,
                                                     qih, keh, Gt_g);
    k_gla_intra<<<dim3(NCHUNK, NH), 256, 0, stream>>>(qih, keh, vbuf, sel, obuf);
    {
        size_t sn = (size_t)NH * 4 * DKH * DVH;
        k_fill_zero<<<dim3((unsigned)((sn + 255) / 256)), 256, 0, stream>>>(Sstate, sn);
    }
    k_gla_inter<<<dim3(NH, 4), 64, 0, stream>>>(qi_g, kd_g, vbuf, Gt_g, sel, Sstate, obuf);
    k_onorm_gate<<<T_SEQ * NH, 256, 0, stream>>>(obuf, gatebuf, wonorm, oh);
    gemm(oh, Wsseo_t, sse_out, T_SEQ, DMODEL, HD);

    // ---- SWA ----
    gemm(xh, Wswaq_t, qs, T_SEQ, HD, DMODEL);
    gemm(xh, Wswak_t, ks, T_SEQ, HD, DMODEL);
    gemm(xh, Wswav_t, vs, T_SEQ, HD, DMODEL);
    {
        int n = T_SEQ * NH * (DKH / 2);
        k_rope<<<(n + 255) / 256, 256, 0, stream>>>(qs, qsh);
        k_rope<<<(n + 255) / 256, 256, 0, stream>>>(ks, ksh);
    }
    cvt(vs, vsh, TH);
    k_swa_flash<<<dim3(T_SEQ / 64, NH), 256, 0, stream>>>(qsh, ksh, vsh, swah);
    gemm(swah, Wswao_t, swa_out, T_SEQ, DMODEL, HD);

    // ---- merge ----
    k_merge<<<T_SEQ, 256, 0, stream>>>(sse_out, swa_out, wsse, wswa, (float*)d_out);
}
